// LIF1d_51135880626581
// MI455X (gfx1250) — matrix operand order repaired and checked by reading, not compiled, not run on hardware
//
#include <hip/hip_runtime.h>

typedef __attribute__((ext_vector_type(16))) _Float16 v16h;
typedef __attribute__((ext_vector_type(8)))  _Float16 v8h;
typedef __attribute__((ext_vector_type(8)))  float    v8f;

#define B_TOT  128
#define D_DIM  512
#define T_DIM  512
#define M_TILE 16          // batch rows per workgroup
#define NT_PW  2           // n-tiles per wave: 32 tiles / 16 waves (512 thr)
#define TC     2           // timestep chunk (float2 staging)

// One-time prep: v (fp32 row-major [K=512][N=512]) -> f16 in WMMA-B layout:
// blob[nt][kt][lane][h], N col = lane%16, K = 8*(lane/16) + (h&7) + 16*(h>>3) in k-tile.
__global__ __launch_bounds__(256) void lif_prep_v(const float* __restrict__ v,
                                                  _Float16* __restrict__ blob) {
  int idx = blockIdx.x * blockDim.x + threadIdx.x;
  if (idx >= D_DIM * D_DIM) return;
  int h  = idx & 15;
  int l  = (idx >> 4) & 31;
  int kt = (idx >> 9) & 15;
  int nt = idx >> 13;
  blob[idx] = (_Float16)v[(kt * 32 + ((l >> 4) << 3) + (h & 7) + ((h >> 3) << 4)) * D_DIM + nt * 16 + (l & 15)];
}

__global__ __launch_bounds__(512) void lif_main(
    const float* __restrict__ x,        // [B][D][T]
    const float* __restrict__ thr,      // [D]
    const float* __restrict__ alpha,    // [D]
    const float* __restrict__ beta,     // [D]
    const _Float16* __restrict__ vblob, // pre-shuffled B operand
    float* __restrict__ out)            // [B][D][T]
{
  // Double-buffered spike tile (next step's A operand), f16: 2 * 16KB LDS.
  __shared__ _Float16 spk[2][M_TILE * D_DIM];

  const int lane  = threadIdx.x & 31;
  const int wave  = threadIdx.x >> 5;
  const int bb    = blockIdx.x * M_TILE;
  const int ncol  = lane & 15;            // C/D fragment: column within tile
  const int mBase = (lane >> 4) << 3;     // C/D fragment: row base (0 or 8)

  // zero both spike buffers (t=0 sees zero previous spikes)
  {
    unsigned int* p = (unsigned int*)&spk[0][0];
    for (int i = threadIdx.x; i < (2 * M_TILE * D_DIM) / 2; i += (int)blockDim.x)
      p[i] = 0u;
  }

  // per-lane column constants (depend only on d = column)
  float thrv[NT_PW], av[NT_PW], bv[NT_PW];
  int dcol[NT_PW];
#pragma unroll
  for (int q = 0; q < NT_PW; ++q) {
    dcol[q] = (wave * NT_PW + q) * 16 + ncol;
    thrv[q] = thr[dcol[q]];
    av[q]   = alpha[dcol[q]];
    bv[q]   = beta[dcol[q]];
  }

  // per-wave B-operand base pointers (blob offset: nt*8192 + kt*512 + lane*16)
  const _Float16* bp0 = vblob + ((size_t)(wave * NT_PW + 0) * 16 * 32 + lane) * 16;
  const _Float16* bp1 = vblob + ((size_t)(wave * NT_PW + 1) * 16 * 32 + lane) * 16;

  // persistent LIF state in WMMA C/D fragment layout
  v8f syn[NT_PW], mem[NT_PW];
#pragma unroll
  for (int q = 0; q < NT_PW; ++q) {
    syn[q] = (v8f){0.f,0.f,0.f,0.f,0.f,0.f,0.f,0.f};
    mem[q] = (v8f){0.f,0.f,0.f,0.f,0.f,0.f,0.f,0.f};
  }

  __syncthreads();

  const int arow = lane & 15;
  const int ksub = (lane >> 4) ? 8 : 0;   // A-fragment K sub-offset

  for (int t0 = 0; t0 < T_DIM; t0 += TC) {
    // stage TC timesteps of input (t contiguous -> float2 loads)
    float xbuf[NT_PW][8][TC];
#pragma unroll
    for (int q = 0; q < NT_PW; ++q)
#pragma unroll
      for (int r = 0; r < 8; ++r) {
        const float2 xv = *(const float2*)(
            x + ((size_t)(bb + mBase + r) * D_DIM + dcol[q]) * T_DIM + t0);
        xbuf[q][r][0] = xv.x; xbuf[q][r][1] = xv.y;
      }

    float sOut[NT_PW][8][TC];

#pragma unroll
    for (int dt = 0; dt < TC; ++dt) {
      const int t    = t0 + dt;
      const int cur  = t & 1;
      const int prev = cur ^ 1;

      // rec = spk(t-1) @ v : M=16, K=512, 2 N-tiles per wave.
      // Manual 2-stage pipeline on the B fragments; DO NOT unroll (spills).
      v8f rec0 = (v8f){0.f,0.f,0.f,0.f,0.f,0.f,0.f,0.f};
      v8f rec1 = (v8f){0.f,0.f,0.f,0.f,0.f,0.f,0.f,0.f};

      v16h b0 = *(const v16h*)(bp0);
      v16h b1 = *(const v16h*)(bp1);

#pragma unroll 1
      for (int kt = 0; kt < 15; ++kt) {
        const int kb = kt * 32 + ksub;
        // A fragment (16-bit 16x32 layout): K = kb..kb+7 and kb+16..kb+23
        const v8h a0 = *(const v8h*)&spk[prev][arow * D_DIM + kb];
        const v8h a1 = *(const v8h*)&spk[prev][arow * D_DIM + kb + 16];
        const v16h a = __builtin_shufflevector(
            a0, a1, 0,1,2,3,4,5,6,7,8,9,10,11,12,13,14,15);
        // prefetch next k-step's B fragments (L2-resident blob)
        const v16h nb0 = *(const v16h*)(bp0 + (size_t)(kt + 1) * 512);
        const v16h nb1 = *(const v16h*)(bp1 + (size_t)(kt + 1) * 512);
        rec0 = __builtin_amdgcn_wmma_f32_16x16x32_f16(
            false, a, false, b0, (short)0, rec0, false, false);
        rec1 = __builtin_amdgcn_wmma_f32_16x16x32_f16(
            false, a, false, b1, (short)0, rec1, false, false);
        b0 = nb0; b1 = nb1;
      }
      { // epilogue: kt = 15
        const int kb = 15 * 32 + ksub;
        const v8h a0 = *(const v8h*)&spk[prev][arow * D_DIM + kb];
        const v8h a1 = *(const v8h*)&spk[prev][arow * D_DIM + kb + 16];
        const v16h a = __builtin_shufflevector(
            a0, a1, 0,1,2,3,4,5,6,7,8,9,10,11,12,13,14,15);
        rec0 = __builtin_amdgcn_wmma_f32_16x16x32_f16(
            false, a, false, b0, (short)0, rec0, false, false);
        rec1 = __builtin_amdgcn_wmma_f32_16x16x32_f16(
            false, a, false, b1, (short)0, rec1, false, false);
      }

      // LIF elementwise update + spike emit (reference order preserved)
#pragma unroll
      for (int q = 0; q < NT_PW; ++q) {
        const v8f rq = (q == 0) ? rec0 : rec1;
#pragma unroll
        for (int r = 0; r < 8; ++r) {
          const float mo = mem[q][r];
          const float so = syn[q][r];
          const float sp = (mo - thrv[q] > 0.f) ? 1.f : 0.f;     // spike from OLD mem
          sOut[q][r][dt] = sp;
          spk[cur][(mBase + r) * D_DIM + dcol[q]] = (_Float16)sp; // A for step t+1
          syn[q][r] = av[q] * so + xbuf[q][r][dt] + rq[r];        // rec uses spk(t-1)
          mem[q][r] = (bv[q] * mo + (1.f - bv[q]) * so) * (1.f - sp); // old syn; hard reset
        }
      }
      __syncthreads();  // spike tile complete before next step's GEMM
    }

    // flush TC timesteps of spikes (t contiguous -> float2 stores)
#pragma unroll
    for (int q = 0; q < NT_PW; ++q)
#pragma unroll
      for (int r = 0; r < 8; ++r) {
        float2 o;
        o.x = sOut[q][r][0]; o.y = sOut[q][r][1];
        *(float2*)(out + ((size_t)(bb + mBase + r) * D_DIM + dcol[q]) * T_DIM + t0) = o;
      }
  }
}

extern "C" void kernel_launch(void* const* d_in, const int* in_sizes, int n_in,
                              void* d_out, int out_size, void* d_ws, size_t ws_size,
                              hipStream_t stream) {
  const float* x     = (const float*)d_in[0];  // input_ [128,512,512]
  const float* thr   = (const float*)d_in[1];  // membrane_threshold [512]
  const float* alpha = (const float*)d_in[2];  // alpha [512]
  const float* beta  = (const float*)d_in[3];  // beta [512]
  const float* v     = (const float*)d_in[4];  // v [512,512]
  float* out         = (float*)d_out;
  _Float16* vblob    = (_Float16*)d_ws;        // 512 KB scratch

  lif_prep_v<<<dim3((D_DIM * D_DIM + 255) / 256), dim3(256), 0, stream>>>(v, vblob);
  lif_main<<<dim3(B_TOT / M_TILE), dim3(512), 0, stream>>>(x, thr, alpha, beta,
                                                           vblob, out);
}
